// DFF_61718680044107
// MI455X (gfx1250) — matrix operand order repaired and checked by reading, not compiled, not run on hardware
//
#include <hip/hip_runtime.h>
#include <stdint.h>

// ---------------------------------------------------------------------------
// Types for CDNA5 WMMA (wave32): D(16x16,f32) = A(16x32,bf16) x B(32x16,bf16) + C
// ---------------------------------------------------------------------------
typedef __attribute__((ext_vector_type(16))) __bf16          v16bf;
typedef __attribute__((ext_vector_type(8)))  uint32_t        v8u;
typedef __attribute__((ext_vector_type(8)))  float           v8f;
typedef __attribute__((ext_vector_type(4)))  int             v4i;

// Async global->LDS copy (CDNA5, tracked by ASYNCcnt). Probed signature:
//   void __builtin_amdgcn_global_load_async_to_lds_b128(
//        as1 v4i* gsrc, as3 v4i* ldst, imm int offset, imm int cpol)
#if defined(__has_builtin)
#  if __has_builtin(__builtin_amdgcn_global_load_async_to_lds_b128)
#    define HAVE_ASYNC_LDS 1
#  endif
#endif
#ifndef HAVE_ASYNC_LDS
#  define HAVE_ASYNC_LDS 0
#endif

#if HAVE_ASYNC_LDS
typedef __attribute__((address_space(1))) v4i* gv4i_p;
typedef __attribute__((address_space(3))) v4i* lv4i_p;
#  define ASYNC_CP16(gsrc, ldst)                                               \
      __builtin_amdgcn_global_load_async_to_lds_b128(                          \
          (gv4i_p)(void*)(gsrc), (lv4i_p)(void*)(ldst), 0, 0)
#  if __has_builtin(__builtin_amdgcn_s_wait_asynccnt)
#    define WAIT_ASYNC() __builtin_amdgcn_s_wait_asynccnt(0)
#  else
#    define WAIT_ASYNC() asm volatile("s_wait_asynccnt 0x0" ::: "memory")
#  endif
#endif

__device__ __forceinline__ unsigned short f2bf(float f) {
    unsigned int u = __float_as_uint(f);
    unsigned int r = u + 0x7FFFu + ((u >> 16) & 1u);   // round-to-nearest-even
    return (unsigned short)(r >> 16);
}

// Two aligned 16B LDS loads -> one v16bf fragment (lowers to ds_load_b128 x2)
__device__ __forceinline__ v16bf frag16(const unsigned short* p0, const unsigned short* p1) {
    uint4 lo = *reinterpret_cast<const uint4*>(p0);
    uint4 hi = *reinterpret_cast<const uint4*>(p1);
    v8u t;
    t[0] = lo.x; t[1] = lo.y; t[2] = lo.z; t[3] = lo.w;
    t[4] = hi.x; t[5] = hi.y; t[6] = hi.z; t[7] = hi.w;
    return __builtin_bit_cast(v16bf, t);
}

// ---------------------------------------------------------------------------
// Generic batched GEMM: C[M,N] = A[M,K] * (TRANSB ? Bt[N,K]^T : B[K,N])
// A,B bf16 (raw u16), C f32 or bf16. Dims multiples of 128 (M,N) / 32 (K).
// Block 256 threads (8 waves); tile 128x128; wave tile 32x64 (2x4 WMMA accum).
// LDS: double-buffered sA[m][k] / sB[n][k], k-stride 40. B always staged
// k-contiguous so both A and B fragments are two ds_load_b128 (ISA layouts).
// Pipeline: issue async loads for tile t+1 before computing tile t.
// ---------------------------------------------------------------------------
template <int TRANSB, int OUTBF>
__global__ void __launch_bounds__(256)
wmma_gemm(const unsigned short* __restrict__ A,
          const unsigned short* __restrict__ Bm,
          void* __restrict__ Cv,
          int M, int N, int K,
          long long strideA, long long strideB, long long strideC)
{
    __shared__ unsigned short sA[2][128 * 40];
    __shared__ unsigned short sB[2][128 * 40];

    const int b   = blockIdx.z;
    const unsigned short* Ab = A  + (size_t)b * strideA;
    const unsigned short* Bb = Bm + (size_t)b * strideB;

    const int m0  = blockIdx.y * 128;
    const int n0  = blockIdx.x * 128;
    const int tid = threadIdx.x;
    const int lane = tid & 31;
    const int wid  = tid >> 5;
    const int wm   = (wid & 3) * 32;   // 4 waves along M
    const int wn   = (wid >> 2) * 64;  // 2 waves along N

    v8f acc[2][4];
#pragma unroll
    for (int i = 0; i < 2; ++i)
#pragma unroll
        for (int j = 0; j < 4; ++j)
            acc[i][j] = (v8f){0.f, 0.f, 0.f, 0.f, 0.f, 0.f, 0.f, 0.f};

    const int mh   = lane & 15;
    const int kb   = (lane >> 4) * 8;    // A-frag K base (ISA 16-bit A layout)
    const int kb16 = (lane >> 4) * 16;   // unused: B frag uses kb, same k order as A

    // ---- stage tile (A and B) for k0 into buffer buf ----
    auto stage = [&](int k0, int buf) {
        // A tile [128 m][32 k]
#if HAVE_ASYNC_LDS
#pragma unroll
        for (int i = 0; i < 2; ++i) {
            int c = tid * 2 + i;                 // 512 x 16B chunks
            int row = c >> 2, seg = c & 3;
            ASYNC_CP16(Ab + (size_t)(m0 + row) * K + k0 + seg * 8,
                       &sA[buf][row * 40 + seg * 8]);
        }
#else
#pragma unroll
        for (int i = 0; i < 8; ++i) {
            int u = tid + i * 256;
            int row = u >> 4, c2 = u & 15;
            uint32_t val = *reinterpret_cast<const uint32_t*>(
                Ab + (size_t)(m0 + row) * K + k0 + c2 * 2);
            *reinterpret_cast<uint32_t*>(&sA[buf][row * 40 + c2 * 2]) = val;
        }
#endif
        // B tile into sB[buf][n][k]
        if (TRANSB) {
#if HAVE_ASYNC_LDS
#pragma unroll
            for (int i = 0; i < 2; ++i) {
                int c = tid * 2 + i;
                int row = c >> 2, seg = c & 3;
                ASYNC_CP16(Bb + (size_t)(n0 + row) * K + k0 + seg * 8,
                           &sB[buf][row * 40 + seg * 8]);
            }
#else
#pragma unroll
            for (int i = 0; i < 8; ++i) {
                int u = tid + i * 256;
                int row = u >> 4, c2 = u & 15;
                uint32_t val = *reinterpret_cast<const uint32_t*>(
                    Bb + (size_t)(n0 + row) * K + k0 + c2 * 2);
                *reinterpret_cast<uint32_t*>(&sB[buf][row * 40 + c2 * 2]) = val;
            }
#endif
        } else {
            // global B[K,N]: 2x2 register transpose while staging.
            int kp = tid >> 4, ng = tid & 15;
            int kk = kp * 2, nn = ng * 8;
            const uint32_t* r0 = reinterpret_cast<const uint32_t*>(
                Bb + (size_t)(k0 + kk) * N + n0 + nn);
            const uint32_t* r1 = reinterpret_cast<const uint32_t*>(
                Bb + (size_t)(k0 + kk + 1) * N + n0 + nn);
#pragma unroll
            for (int i = 0; i < 4; ++i) {
                uint32_t a  = r0[i];
                uint32_t b2 = r1[i];
                uint32_t lo = (a & 0xffffu) | (b2 << 16);          // {B[k][n],   B[k+1][n]}
                uint32_t hi = (a >> 16)    | (b2 & 0xffff0000u);   // {B[k][n+1], B[k+1][n+1]}
                *reinterpret_cast<uint32_t*>(&sB[buf][(nn + 2 * i)     * 40 + kk]) = lo;
                *reinterpret_cast<uint32_t*>(&sB[buf][(nn + 2 * i + 1) * 40 + kk]) = hi;
            }
        }
    };

    const int T = K >> 5;               // K / 32 tiles
    stage(0, 0);

    for (int t = 0; t < T; ++t) {
        const int cur = t & 1;
#if HAVE_ASYNC_LDS
        WAIT_ASYNC();                   // this wave's async stores to LDS done
#endif
        __syncthreads();                // all waves' tile-t data visible;
                                        // buffer 1-cur fully consumed by all
        if (t + 1 < T)
            stage((t + 1) << 5, 1 - cur);   // prefetch overlaps compute below

        // ---- fragments: two ds_load_b128 each ----
        v16bf av[2], bv[4];
#pragma unroll
        for (int ti = 0; ti < 2; ++ti) {
            const unsigned short* base = &sA[cur][(wm + ti * 16 + mh) * 40];
            av[ti] = frag16(base + kb, base + 16 + kb);   // K = kb..kb+7, 16+kb..
        }
#pragma unroll
        for (int tj = 0; tj < 4; ++tj) {
            const unsigned short* base = &sB[cur][(wn + tj * 16 + mh) * 40];
            bv[tj] = frag16(base + kb, base + 16 + kb);    // K = kb..kb+7, 16+kb..
        }

#pragma unroll
        for (int ti = 0; ti < 2; ++ti)
#pragma unroll
            for (int tj = 0; tj < 4; ++tj)
                acc[ti][tj] = __builtin_amdgcn_wmma_f32_16x16x32_bf16(
                    false, av[ti], false, bv[tj], (short)0, acc[ti][tj], false, false);
    }

    // ---- writeback (C layout: VGPR r -> M = r + 8*(lane>>4), N = lane&15) ----
    const int rh = (lane >> 4) * 8;
#pragma unroll
    for (int ti = 0; ti < 2; ++ti)
#pragma unroll
        for (int tj = 0; tj < 4; ++tj)
#pragma unroll
            for (int r = 0; r < 8; ++r) {
                int m = m0 + wm + ti * 16 + r + rh;
                int n = n0 + wn + tj * 16 + mh;
                float v = acc[ti][tj][r];
                size_t off = (size_t)b * strideC + (size_t)m * N + n;
                if (OUTBF) ((unsigned short*)Cv)[off] = f2bf(v);
                else       ((float*)Cv)[off] = v;
            }
}

// ---------------------------------------------------------------------------
// Elementwise / reduction kernels
// ---------------------------------------------------------------------------
// [B,C,N] f32 -> [B,N,C] bf16
__global__ void k_transpose_cvt(const float* __restrict__ in,
                                unsigned short* __restrict__ out, int C, int N)
{
    int b = blockIdx.z;
    int idx = blockIdx.x * blockDim.x + threadIdx.x;
    if (idx >= N * C) return;
    int n = idx / C, c = idx % C;
    out[(size_t)b * N * C + idx] = f2bf(in[(size_t)b * C * N + (size_t)c * N + n]);
}

__global__ void k_cvt_bf16(const float* __restrict__ in, unsigned short* __restrict__ out, int n)
{
    int i = blockIdx.x * blockDim.x + threadIdx.x;
    if (i < n) out[i] = f2bf(in[i]);
}

// per-batch sum / sumsq of S (for InstanceNorm variance); stats must be zeroed
__global__ void k_inst_stats(const float* __restrict__ S, double* stats, int perBatch)
{
    __shared__ float rs[256], rs2[256];
    int b = blockIdx.y;
    int chunk = perBatch / gridDim.x;
    const float* p = S + (size_t)b * perBatch + (size_t)blockIdx.x * chunk;
    float s = 0.f, s2 = 0.f;
    for (int i = threadIdx.x; i < chunk; i += 256) { float v = p[i]; s += v; s2 += v * v; }
    rs[threadIdx.x] = s; rs2[threadIdx.x] = s2; __syncthreads();
    for (int o = 128; o > 0; o >>= 1) {
        if (threadIdx.x < o) { rs[threadIdx.x] += rs[threadIdx.x + o]; rs2[threadIdx.x] += rs2[threadIdx.x + o]; }
        __syncthreads();
    }
    if (threadIdx.x == 0) {
        atomicAdd(&stats[b * 2 + 0], (double)rs[0]);
        atomicAdd(&stats[b * 2 + 1], (double)rs2[0]);
    }
}

// softmax((S - mean) * r) == softmax(S * r)  (mean shift cancels row-wise)
__global__ void __launch_bounds__(256)
k_softmax(const float* __restrict__ S, unsigned short* __restrict__ A,
          const double* __restrict__ stats, int L, int rowsPerBatch, float cnt)
{
    __shared__ float rowbuf[4096];
    __shared__ float red[256];
    int row = blockIdx.x;
    int b = row / rowsPerBatch;
    double mean = stats[b * 2 + 0] / (double)cnt;
    double var  = stats[b * 2 + 1] / (double)cnt - mean * mean;
    float r = rsqrtf((float)var + 1e-5f);
    const float* p = S + (size_t)row * L;

    float mx = -3.4e38f;
    for (int i = threadIdx.x; i < L; i += 256) { float v = p[i] * r; rowbuf[i] = v; mx = fmaxf(mx, v); }
    red[threadIdx.x] = mx; __syncthreads();
    for (int o = 128; o > 0; o >>= 1) {
        if (threadIdx.x < o) red[threadIdx.x] = fmaxf(red[threadIdx.x], red[threadIdx.x + o]);
        __syncthreads();
    }
    mx = red[0]; __syncthreads();

    float s = 0.f;
    for (int i = threadIdx.x; i < L; i += 256) { float e = __expf(rowbuf[i] - mx); rowbuf[i] = e; s += e; }
    red[threadIdx.x] = s; __syncthreads();
    for (int o = 128; o > 0; o >>= 1) {
        if (threadIdx.x < o) red[threadIdx.x] += red[threadIdx.x + o];
        __syncthreads();
    }
    float inv = 1.f / red[0];
    unsigned short* q = A + (size_t)row * L;
    for (int i = threadIdx.x; i < L; i += 256) q[i] = f2bf(rowbuf[i] * inv);
}

// per-channel mean / rstd over rows (BatchNorm, training stats). Conv bias
// cancels under BN mean subtraction so it is never added.
__global__ void k_bn_stats(const float* __restrict__ Z, float* __restrict__ cs, int rows, int C)
{
    __shared__ float rs[256], rs2[256];
    int c = blockIdx.x;
    float s = 0.f, s2 = 0.f;
    for (int r = threadIdx.x; r < rows; r += 256) { float v = Z[(size_t)r * C + c]; s += v; s2 += v * v; }
    rs[threadIdx.x] = s; rs2[threadIdx.x] = s2; __syncthreads();
    for (int o = 128; o > 0; o >>= 1) {
        if (threadIdx.x < o) { rs[threadIdx.x] += rs[threadIdx.x + o]; rs2[threadIdx.x] += rs2[threadIdx.x + o]; }
        __syncthreads();
    }
    if (threadIdx.x == 0) {
        float mean = rs[0] / (float)rows;
        float var  = rs2[0] / (float)rows - mean * mean;
        cs[2 * c + 0] = mean;
        cs[2 * c + 1] = rsqrtf(var + 1e-5f);
    }
}

// out[b][o][n] = relu((Z[b][n][o] - mu_o) * rstd_o * g_o + beta_o)
__global__ void k_bn_apply(const float* __restrict__ Z, const float* __restrict__ cs,
                           const float* __restrict__ g, const float* __restrict__ beta,
                           float* __restrict__ out, int N, int C, int total)
{
    int idx = blockIdx.x * blockDim.x + threadIdx.x;
    if (idx >= total) return;
    int b = idx / (N * C);
    int rem = idx - b * N * C;
    int o = rem / N, n = rem - o * N;
    float z = Z[(size_t)b * N * C + (size_t)n * C + o];
    float y = (z - cs[2 * o]) * cs[2 * o + 1] * g[o] + beta[o];
    out[idx] = fmaxf(y, 0.f);
}

// ---------------------------------------------------------------------------
// Host launcher
// ---------------------------------------------------------------------------
static inline void launch_gemm(int transB, int outBF,
                               const unsigned short* A, const unsigned short* B, void* C,
                               int M, int N, int K,
                               long long sa, long long sb, long long sc,
                               int batch, hipStream_t st)
{
    dim3 g(N / 128, M / 128, batch), blk(256);
    if (!transB && !outBF)      wmma_gemm<0, 0><<<g, blk, 0, st>>>(A, B, C, M, N, K, sa, sb, sc);
    else if (!transB &&  outBF) wmma_gemm<0, 1><<<g, blk, 0, st>>>(A, B, C, M, N, K, sa, sb, sc);
    else if ( transB && !outBF) wmma_gemm<1, 0><<<g, blk, 0, st>>>(A, B, C, M, N, K, sa, sb, sc);
    else                        wmma_gemm<1, 1><<<g, blk, 0, st>>>(A, B, C, M, N, K, sa, sb, sc);
}

extern "C" void kernel_launch(void* const* d_in, const int* in_sizes, int n_in,
                              void* d_out, int out_size, void* d_ws, size_t ws_size,
                              hipStream_t stream)
{
    (void)in_sizes; (void)n_in; (void)out_size;
    const int B = 4, DL = 256, DS = 512, NQ = 4096, NK = 1024;

    const float* q       = (const float*)d_in[0];
    const float* k       = (const float*)d_in[1];
    const float* Wq1     = (const float*)d_in[2];
    const float* Wk1     = (const float*)d_in[3];
    const float* Wv1     = (const float*)d_in[4];
    const float* Wo1     = (const float*)d_in[5];
    const float* Wq2     = (const float*)d_in[6];
    const float* Wk2     = (const float*)d_in[7];
    const float* Wv2     = (const float*)d_in[8];
    const float* Wo2     = (const float*)d_in[9];
    const float* conv1_w = (const float*)d_in[10];
    const float* bn1_g   = (const float*)d_in[12];
    const float* bn1_b   = (const float*)d_in[13];
    const float* conv2_w = (const float*)d_in[14];
    const float* bn2_g   = (const float*)d_in[16];
    const float* bn2_b   = (const float*)d_in[17];
    float* out = (float*)d_out;

    // ---- workspace arena (~192 MB) ----
    char* base = (char*)d_ws;
    size_t off = 0;
    auto alloc = [&](size_t bytes) { void* p = base + off; off = (off + bytes + 255) & ~(size_t)255; return p; };
    unsigned short* qf   = (unsigned short*)alloc((size_t)B * NQ * DL * 2);   // [B,Nq,DL] bf16
    unsigned short* kf   = (unsigned short*)alloc((size_t)B * NK * DS * 2);   // [B,Nk,DS] bf16
    unsigned short* wq1b = (unsigned short*)alloc((size_t)DL * DL * 2);
    unsigned short* wk1b = (unsigned short*)alloc((size_t)DS * DL * 2);
    unsigned short* wv1b = (unsigned short*)alloc((size_t)DS * DL * 2);
    unsigned short* wo1b = (unsigned short*)alloc((size_t)DL * DL * 2);
    unsigned short* wq2b = (unsigned short*)alloc((size_t)DS * DS * 2);
    unsigned short* wk2b = (unsigned short*)alloc((size_t)DL * DS * 2);
    unsigned short* wv2b = (unsigned short*)alloc((size_t)DL * DS * 2);
    unsigned short* wo2b = (unsigned short*)alloc((size_t)DS * DS * 2);
    unsigned short* cv1b = (unsigned short*)alloc((size_t)DL * DL * 2);
    unsigned short* cv2b = (unsigned short*)alloc((size_t)DS * DS * 2);
    unsigned short* Pq   = (unsigned short*)alloc((size_t)B * NQ * DL * 2);   // max(Q1,Q2)
    unsigned short* Pk   = (unsigned short*)alloc((size_t)B * NQ * DS * 2);   // max(K1,K2)
    unsigned short* Pv   = (unsigned short*)alloc((size_t)B * NQ * DS * 2);
    float*          S    = (float*)alloc((size_t)B * NQ * NK * 4);            // scores, f32
    unsigned short* Am   = (unsigned short*)alloc((size_t)B * NQ * NK * 2);   // softmax out, bf16
    unsigned short* O    = (unsigned short*)alloc((size_t)B * NQ * DL * 2);
    unsigned short* X    = (unsigned short*)alloc((size_t)B * NQ * DL * 2);
    float*          Z    = (float*)alloc((size_t)B * NQ * DL * 4);            // conv out, f32
    double*         ist  = (double*)alloc(8 * sizeof(double));
    float*          cst  = (float*)alloc(2 * DS * sizeof(float));
    (void)ws_size;

    dim3 b256(256);

    // ---- prologue: transposes + weight conversion ----
    k_transpose_cvt<<<dim3((NQ * DL) / 256, 1, B), b256, 0, stream>>>(q, qf, DL, NQ);
    k_transpose_cvt<<<dim3((NK * DS) / 256, 1, B), b256, 0, stream>>>(k, kf, DS, NK);
    k_cvt_bf16<<<dim3(DL * DL / 256), b256, 0, stream>>>(Wq1, wq1b, DL * DL);
    k_cvt_bf16<<<dim3(DS * DL / 256), b256, 0, stream>>>(Wk1, wk1b, DS * DL);
    k_cvt_bf16<<<dim3(DS * DL / 256), b256, 0, stream>>>(Wv1, wv1b, DS * DL);
    k_cvt_bf16<<<dim3(DL * DL / 256), b256, 0, stream>>>(Wo1, wo1b, DL * DL);
    k_cvt_bf16<<<dim3(DS * DS / 256), b256, 0, stream>>>(Wq2, wq2b, DS * DS);
    k_cvt_bf16<<<dim3(DL * DS / 256), b256, 0, stream>>>(Wk2, wk2b, DL * DS);
    k_cvt_bf16<<<dim3(DL * DS / 256), b256, 0, stream>>>(Wv2, wv2b, DL * DS);
    k_cvt_bf16<<<dim3(DS * DS / 256), b256, 0, stream>>>(Wo2, wo2b, DS * DS);
    k_cvt_bf16<<<dim3(DL * DL / 256), b256, 0, stream>>>(conv1_w, cv1b, DL * DL);
    k_cvt_bf16<<<dim3(DS * DS / 256), b256, 0, stream>>>(conv2_w, cv2b, DS * DS);

    // ======================= Phase 1: cross_atten1 (q<-k) ====================
    launch_gemm(0, 1, qf, wq1b, Pq, NQ, DL, DL, (long long)NQ * DL, 0, (long long)NQ * DL, B, stream); // Q1
    launch_gemm(0, 1, kf, wk1b, Pk, NK, DL, DS, (long long)NK * DS, 0, (long long)NK * DL, B, stream); // K1
    launch_gemm(0, 1, kf, wv1b, Pv, NK, DL, DS, (long long)NK * DS, 0, (long long)NK * DL, B, stream); // V1
    // S1 = Q1 @ K1^T : [B,4096,1024], f32
    launch_gemm(1, 0, Pq, Pk, S, NQ, NK, DL, (long long)NQ * DL, (long long)NK * DL, (long long)NQ * NK, B, stream);
    (void)hipMemsetAsync(ist, 0, 8 * sizeof(double), stream);
    k_inst_stats<<<dim3(1024, B), b256, 0, stream>>>(S, ist, NQ * NK);
    k_softmax<<<dim3(B * NQ), b256, 0, stream>>>(S, Am, ist, NK, NQ, (float)(NQ * NK));
    // O1 = A1 @ V1 : [B,4096,256]
    launch_gemm(0, 1, Am, Pv, O, NQ, DL, NK, (long long)NQ * NK, (long long)NK * DL, (long long)NQ * DL, B, stream);
    launch_gemm(0, 1, O, wo1b, X, NQ, DL, DL, (long long)NQ * DL, 0, (long long)NQ * DL, B, stream);   // X1
    // Z1[b,n,o] = X1 @ conv1_w^T (conv bias cancels in BN)
    launch_gemm(1, 0, X, cv1b, Z, NQ, DL, DL, (long long)NQ * DL, 0, (long long)NQ * DL, B, stream);
    k_bn_stats<<<dim3(DL), b256, 0, stream>>>(Z, cst, B * NQ, DL);
    k_bn_apply<<<dim3((B * NQ * DL) / 256), b256, 0, stream>>>(Z, cst, bn1_g, bn1_b, out, NQ, DL, B * NQ * DL);

    // ======================= Phase 2: cross_atten2 (k<-q) ====================
    launch_gemm(0, 1, kf, wq2b, Pq, NK, DS, DS, (long long)NK * DS, 0, (long long)NK * DS, B, stream); // Q2
    launch_gemm(0, 1, qf, wk2b, Pk, NQ, DS, DL, (long long)NQ * DL, 0, (long long)NQ * DS, B, stream); // K2
    launch_gemm(0, 1, qf, wv2b, Pv, NQ, DS, DL, (long long)NQ * DL, 0, (long long)NQ * DS, B, stream); // V2
    // S2 = Q2 @ K2^T : [B,1024,4096], f32
    launch_gemm(1, 0, Pq, Pk, S, NK, NQ, DS, (long long)NK * DS, (long long)NQ * DS, (long long)NK * NQ, B, stream);
    (void)hipMemsetAsync(ist, 0, 8 * sizeof(double), stream);
    k_inst_stats<<<dim3(1024, B), b256, 0, stream>>>(S, ist, NK * NQ);
    k_softmax<<<dim3(B * NK), b256, 0, stream>>>(S, Am, ist, NQ, NK, (float)(NK * NQ));
    // O2 = A2 @ V2 : [B,1024,512]
    launch_gemm(0, 1, Am, Pv, O, NK, DS, NQ, (long long)NK * NQ, (long long)NQ * DS, (long long)NK * DS, B, stream);
    launch_gemm(0, 1, O, wo2b, X, NK, DS, DS, (long long)NK * DS, 0, (long long)NK * DS, B, stream);   // X2
    launch_gemm(1, 0, X, cv2b, Z, NK, DS, DS, (long long)NK * DS, 0, (long long)NK * DS, B, stream);   // Z2
    k_bn_stats<<<dim3(DS), b256, 0, stream>>>(Z, cst, B * NK, DS);
    k_bn_apply<<<dim3((B * NK * DS) / 256), b256, 0, stream>>>(Z, cst, bn2_g, bn2_b,
                                                               out + (size_t)B * NQ * DL, NK, DS, B * NK * DS);
}
